// AttentionLayer_57183194579071
// MI455X (gfx1250) — matrix operand order repaired and checked by reading, not compiled, not run on hardware
//
#include <hip/hip_runtime.h>
#include <hip/hip_bf16.h>

// Problem constants (match reference)
#define B_ 2
#define L_ 2048
#define D_ 1024
#define H_ 16
#define E_ 64
#define N_ (B_*L_)   // 4096 rows

typedef __bf16 bf16;
typedef __attribute__((ext_vector_type(16))) __bf16 v16bf;
typedef __attribute__((ext_vector_type(8)))  __bf16 v8bf;
typedef __attribute__((ext_vector_type(8)))  float  v8f;

// ---------------------------------------------------------------------------
// CDNA5 async copy: global -> LDS, 16 bytes per lane, tracked by ASYNCcnt.
// LDS byte address = low 32 bits of the generic pointer to shared memory.
// ---------------------------------------------------------------------------
#define ASYNC_CP_B128(ldsptr, gptr)                                          \
  asm volatile("global_load_async_to_lds_b128 %0, %1, off"                   \
               :: "v"((unsigned)(uintptr_t)(ldsptr)),                        \
                  "v"((unsigned long long)(uintptr_t)(gptr))                 \
               : "memory")

#define WAIT_ASYNC_0() asm volatile("s_wait_asynccnt 0" ::: "memory")
#define WAIT_ASYNC_2() asm volatile("s_wait_asynccnt 2" ::: "memory")
#define WAIT_ASYNC_4() asm volatile("s_wait_asynccnt 4" ::: "memory")
#define WAIT_DS_0()    asm volatile("s_wait_dscnt 0"    ::: "memory")

// ---------------------------------------------------------------------------
// WMMA fragment helpers (wave32). Layouts per CDNA5 ISA 7.12.2.
// ---------------------------------------------------------------------------

// A-matrix 16x32 bf16 fragment from row-major memory (leading dim ld, elems).
// lane l: m=l&15, koff=(l>>4)*8 ; elems 0..7 = K[koff..koff+7], 8..15 = K[16+koff..]
static __device__ __forceinline__ v16bf load_a_frag(const bf16* base, int ld) {
  const int l = threadIdx.x & 31;
  const int m = l & 15;
  const int koff = (l >> 4) * 8;
  const v8bf lo = *(const v8bf*)(base + (size_t)m * ld + koff);
  const v8bf hi = *(const v8bf*)(base + (size_t)m * ld + 16 + koff);
  v16bf a;
#pragma unroll
  for (int i = 0; i < 8; ++i) { a[i] = lo[i]; a[8 + i] = hi[i]; }
  return a;
}

// B-matrix 32x16 bf16 fragment. Source is K-contiguous ("transposed"):
// element (k, n) lives at baseT[n*ldt + k].
// lane l: n=l&15, kb=(l>>4)*8 ; elems 0..7 = (kb+i, n), 8..15 = (16+kb+i, n)
static __device__ __forceinline__ v16bf load_b_frag(const bf16* baseT, int ldt) {
  const int l = threadIdx.x & 31;
  const int n = l & 15;
  const int kb = (l >> 4) * 8;
  const v8bf lo = *(const v8bf*)(baseT + (size_t)n * ldt + kb);
  const v8bf hi = *(const v8bf*)(baseT + (size_t)n * ldt + 16 + kb);
  v16bf b;
#pragma unroll
  for (int i = 0; i < 8; ++i) { b[i] = lo[i]; b[8 + i] = hi[i]; }
  return b;
}

static __device__ __forceinline__ v8f wmma_bf16(v16bf a, v16bf b, v8f c) {
  return __builtin_amdgcn_wmma_f32_16x16x32_bf16(
      /*neg_a=*/false, a, /*neg_b=*/false, b,
      /*c_mod=*/(short)0, c, /*reuse_a=*/false, /*reuse_b=*/false);
}

// ---------------------------------------------------------------------------
// fp32 -> bf16 convert kernels
// ---------------------------------------------------------------------------
__global__ __launch_bounds__(256) void k_cvt_bf16(const float* __restrict__ src,
                                                  bf16* __restrict__ dst, int n) {
  int i = blockIdx.x * blockDim.x + threadIdx.x;
  if (i < n) dst[i] = (bf16)src[i];
}

// W[rows, cols] fp32 row-major  ->  WT[cols, rows] bf16 (K-contiguous columns)
__global__ __launch_bounds__(256) void k_cvt_transpose(const float* __restrict__ W,
                                                       bf16* __restrict__ WT,
                                                       int rows, int cols) {
  int i = blockIdx.x * blockDim.x + threadIdx.x;   // i enumerates (c, r), r fastest
  if (i >= rows * cols) return;
  int c = i / rows;
  int r = i - c * rows;
  WT[(size_t)c * rows + r] = (bf16)W[(size_t)r * cols + c];
}

// ---------------------------------------------------------------------------
// WMMA GEMM with double-buffered async-to-LDS staging.
// Workgroup (8 waves) computes a 128x128 tile of C[M, Ncol] = A x W + bias,
// A bf16 [M,K] row-major, W given as WT[Ncol, K] bf16 (K-contiguous).
// Per K-step: stage A(128x32) + B(128x32) into LDS (4 async b128 per lane),
// prefetch next stage while computing current (s_wait_asynccnt 4).
// Wave (wid&3, wid>>2) computes 32 rows x 64 cols: 2 A frags x 4 B frags
// loaded up front, then 8 back-to-back WMMAs (full register reuse).
// mode 0: bf16 row-major | 1: bf16 [B,H,L,E] | 2: bf16 [B,H,E,L]
// mode 3: fp32 out with fused gating: out = g*(acc+bias) + (1-g)*vglob[b]
// ---------------------------------------------------------------------------
#define LDA_ 40   // padded stride (halves) for 32-wide K panels
__global__ __launch_bounds__(256)
void k_gemm(const bf16* __restrict__ A, const bf16* __restrict__ WT,
            const float* __restrict__ bias, void* __restrict__ out,
            int M, int K, int Ncol, int mode,
            const float* __restrict__ gate, const float* __restrict__ vglob) {
  __shared__ bf16 AL[2][128 * LDA_];
  __shared__ bf16 BL[2][128 * LDA_];

  const int tiles_n = Ncol >> 7;                 // 128-wide tiles
  const int bm = blockIdx.x / tiles_n;
  const int bn = blockIdx.x - bm * tiles_n;
  const int row0 = bm << 7, col0 = bn << 7;

  const int wid  = threadIdx.x >> 5;
  const int lane = threadIdx.x & 31;
  const int wrow = row0 + (wid & 3) * 32;        // this wave's 32 rows
  const int wcol = col0 + (wid >> 2) * 64;       // this wave's 64 cols

  const int t = threadIdx.x;
  // stage(buf, k0): A rows [row0,row0+128) & B cols [col0,col0+128), k 32 wide.
  // 128 x 32 halves = 512 b128 xfers per matrix -> 2 per thread per matrix.
  auto stage = [&](int buf, int k0) {
#pragma unroll
    for (int j = 0; j < 2; ++j) {
      const int idx = t + j * 256;
      const int r  = idx >> 2;
      const int kk = (idx & 3) * 8;
      ASYNC_CP_B128(&AL[buf][r * LDA_ + kk], A  + (size_t)(row0 + r) * K + k0 + kk);
      ASYNC_CP_B128(&BL[buf][r * LDA_ + kk], WT + (size_t)(col0 + r) * K + k0 + kk);
    }
  };

  v8f acc[2][4] = {};
  stage(0, 0);
  for (int k0 = 0; k0 < K; k0 += 32) {
    const int cur = (k0 >> 5) & 1;
    if (k0 + 32 < K) { stage(cur ^ 1, k0 + 32); WAIT_ASYNC_4(); }
    else             { WAIT_ASYNC_0(); }
    __syncthreads();

    // load all fragments first so ds_loads clause together, then 8 WMMAs
    const v16bf a0 = load_a_frag(&AL[cur][((wid & 3) * 32)      * LDA_], LDA_);
    const v16bf a1 = load_a_frag(&AL[cur][((wid & 3) * 32 + 16) * LDA_], LDA_);
    v16bf b[4];
#pragma unroll
    for (int s = 0; s < 4; ++s)
      b[s] = load_b_frag(&BL[cur][((wid >> 2) * 64 + s * 16) * LDA_], LDA_);
#pragma unroll
    for (int s = 0; s < 4; ++s) acc[0][s] = wmma_bf16(a0, b[s], acc[0][s]);
#pragma unroll
    for (int s = 0; s < 4; ++s) acc[1][s] = wmma_bf16(a1, b[s], acc[1][s]);
    __syncthreads();                             // all reads done before overwrite
  }

  const int nlo   = lane & 15;
  const int mbase = (lane >> 4) * 8;
#pragma unroll
  for (int h = 0; h < 2; ++h)
#pragma unroll
    for (int s = 0; s < 4; ++s) {
      const int col = wcol + s * 16 + nlo;
      const float bz = bias ? bias[col] : 0.0f;
#pragma unroll
      for (int r = 0; r < 8; ++r) {
        const int row = wrow + h * 16 + mbase + r;
        const float v = acc[h][s][r] + bz;
        if (mode == 0) {
          ((bf16*)out)[(size_t)row * Ncol + col] = (bf16)v;
        } else if (mode == 1) {          // [B,H,L,E]
          const int bb = row / L_, ll = row - bb * L_;
          const int hh = col >> 6, ee = col & 63;
          ((bf16*)out)[(((size_t)(bb * H_ + hh) * L_) + ll) * E_ + ee] = (bf16)v;
        } else if (mode == 2) {          // [B,H,E,L]
          const int bb = row / L_, ll = row - bb * L_;
          const int hh = col >> 6, ee = col & 63;
          ((bf16*)out)[(((size_t)(bb * H_ + hh) * E_) + ee) * L_ + ll] = (bf16)v;
        } else {                         // fused gated blend, fp32 out
          const int bb = row / L_;
          const float g = gate[row];
          const float vg = vglob[(size_t)bb * D_ + col];
          ((float*)out)[(size_t)row * Ncol + col] = g * v + (1.0f - g) * vg;
        }
      }
    }
}

// ---------------------------------------------------------------------------
// Flash attention, block-cooperative. One block = one (b,h) x 128 q-rows
// (8 waves x 16 rows). Per s-step (32 keys): K tile (32 x 64) and Vt tile
// (64 x 32) staged once per block into LDS via async copies (2 per lane),
// double-buffered; all 8 waves consume them. Online softmax in fp32; P is
// re-packed C-layout -> A-layout through per-wave LDS scratch.
// Q,K: [B,H,L,E] bf16 ; Vt: [B,H,E,L] bf16 ; O: [B,L,H,E] bf16
// ---------------------------------------------------------------------------
#define LDK_ 72   // K tile stride (halves): 64 e + pad
__global__ __launch_bounds__(256)
void k_flash(const bf16* __restrict__ Q, const bf16* __restrict__ Km,
             const bf16* __restrict__ Vt, bf16* __restrict__ O) {
  __shared__ bf16 KL[2][32 * LDK_];
  __shared__ bf16 VL[2][64 * LDA_];
  __shared__ bf16 PB[8][16 * LDA_];

  const int QB = L_ / 128;                   // 16 q-blocks per head
  const int bh = blockIdx.x / QB;            // fused (b*H + h)
  const int qb = blockIdx.x - bh * QB;

  const int wid  = threadIdx.x >> 5;
  const int lane = threadIdx.x & 31;
  const int q0   = qb * 128 + wid * 16;      // this wave's 16 q rows

  const bf16* Qh = Q  + ((size_t)bh * L_ + q0) * E_;
  const bf16* Kh = Km + (size_t)bh * L_ * E_;
  const bf16* Vh = Vt + (size_t)bh * E_ * L_;
  bf16* pl = PB[wid];

  const int t  = threadIdx.x;
  const int ks = t >> 3, ke = (t & 7) * 8;   // K tile: 32 rows x 8 xfers
  const int ve = t >> 2, vs = (t & 3) * 8;   // V tile: 64 rows x 4 xfers
  auto stage = [&](int buf, int s0) {
    ASYNC_CP_B128(&KL[buf][ks * LDK_ + ke], Kh + (size_t)(s0 + ks) * E_ + ke);
    ASYNC_CP_B128(&VL[buf][ve * LDA_ + vs], Vh + (size_t)ve * L_ + s0 + vs);
  };

  const v16bf qa0 = load_a_frag(Qh, E_);       // e = 0..31
  const v16bf qa1 = load_a_frag(Qh + 32, E_);  // e = 32..63

  v8f oacc[4] = {};
  float mrow[8], lrow[8];
#pragma unroll
  for (int r = 0; r < 8; ++r) { mrow[r] = -3.0e38f; lrow[r] = 0.0f; }

  const float scale = 0.125f;                // 1/sqrt(E)
  const int mbase = (lane >> 4) * 8;
  const int nlo   = lane & 15;

  stage(0, 0);
  for (int s0 = 0; s0 < L_; s0 += 32) {
    const int cur = (s0 >> 5) & 1;
    if (s0 + 32 < L_) { stage(cur ^ 1, s0 + 32); WAIT_ASYNC_2(); }
    else              { WAIT_ASYNC_0(); }
    __syncthreads();

    // ---- scores: two 16x16 tiles over K-dim e (2 WMMAs each) ----
    v8f sc[2] = {};
#pragma unroll
    for (int c = 0; c < 2; ++c) {
      v16bf kb0 = load_b_frag(&KL[cur][(c * 16) * LDK_], LDK_);
      v16bf kb1 = load_b_frag(&KL[cur][(c * 16) * LDK_ + 32], LDK_);
      sc[c] = wmma_bf16(qa0, kb0, sc[c]);
      sc[c] = wmma_bf16(qa1, kb1, sc[c]);
#pragma unroll
      for (int r = 0; r < 8; ++r) sc[c][r] *= scale;
    }

    // ---- online softmax row statistics (rows live on 16-lane halves) ----
    float mnew[8], alpha[8];
#pragma unroll
    for (int r = 0; r < 8; ++r) {
      float v = fmaxf(sc[0][r], sc[1][r]);
      v = fmaxf(v, __shfl_xor(v, 1, 32));
      v = fmaxf(v, __shfl_xor(v, 2, 32));
      v = fmaxf(v, __shfl_xor(v, 4, 32));
      v = fmaxf(v, __shfl_xor(v, 8, 32));
      mnew[r]  = fmaxf(mrow[r], v);
      alpha[r] = __expf(mrow[r] - mnew[r]);
      mrow[r]  = mnew[r];
    }

    // ---- P = exp(S - m) -> per-wave LDS scratch, accumulate row sums ----
#pragma unroll
    for (int r = 0; r < 8; ++r) {
      const float p0 = __expf(sc[0][r] - mnew[r]);
      const float p1 = __expf(sc[1][r] - mnew[r]);
      pl[(mbase + r) * LDA_ + nlo]      = (bf16)p0;
      pl[(mbase + r) * LDA_ + 16 + nlo] = (bf16)p1;
      float ps = p0 + p1;
      ps += __shfl_xor(ps, 1, 32);
      ps += __shfl_xor(ps, 2, 32);
      ps += __shfl_xor(ps, 4, 32);
      ps += __shfl_xor(ps, 8, 32);
      lrow[r] = lrow[r] * alpha[r] + ps;
    }
    WAIT_DS_0();                               // same-wave LDS bounce

    // ---- rescale accumulators, then O += P x V ----
#pragma unroll
    for (int en = 0; en < 4; ++en)
#pragma unroll
      for (int r = 0; r < 8; ++r) oacc[en][r] *= alpha[r];

    const v16bf pa = load_a_frag(pl, LDA_);    // P as 16x32 A fragment
#pragma unroll
    for (int en = 0; en < 4; ++en) {
      v16bf vb = load_b_frag(&VL[cur][(en * 16) * LDA_], LDA_);
      oacc[en] = wmma_bf16(pa, vb, oacc[en]);
    }
    __syncthreads();                           // K/V reads done before overwrite
  }

  // ---- normalize and write O[b, l, h, e] ----
  const int bb = bh / H_, hh = bh - bb * H_;
#pragma unroll
  for (int en = 0; en < 4; ++en)
#pragma unroll
    for (int r = 0; r < 8; ++r) {
      const int ll = q0 + mbase + r;
      const int ee = en * 16 + nlo;
      const float v = oacc[en][r] / lrow[r];
      O[(((size_t)(bb * L_ + ll) * H_) + hh) * E_ + ee] = (bf16)v;
    }
}

// ---------------------------------------------------------------------------
// Small fp32 epilogue helpers (negligible FLOPs, keep full precision)
// ---------------------------------------------------------------------------
__global__ __launch_bounds__(256) void k_colmean(const float* __restrict__ x,
                                                 float* __restrict__ g) {
  int i = blockIdx.x * blockDim.x + threadIdx.x;   // i = b*D + d
  if (i >= B_ * D_) return;
  const int bb = i / D_, dd = i - bb * D_;
  float s = 0.0f;
  for (int l = 0; l < L_; ++l) s += x[((size_t)bb * L_ + l) * D_ + dd];
  g[i] = s * (1.0f / (float)L_);
}

__global__ __launch_bounds__(256) void k_vglobal(const float* __restrict__ g,
                                                 const float* __restrict__ Wg,
                                                 const float* __restrict__ bg,
                                                 float* __restrict__ vg) {
  int i = blockIdx.x * blockDim.x + threadIdx.x;   // i = b*D + c
  if (i >= B_ * D_) return;
  const int bb = i / D_, cc = i - bb * D_;
  float s = bg[cc];
  for (int d = 0; d < D_; ++d) s += g[(size_t)bb * D_ + d] * Wg[(size_t)d * D_ + cc];
  vg[i] = s;
}

__global__ __launch_bounds__(256) void k_gate(const float* __restrict__ x,
                                              const float* __restrict__ Wf,
                                              const float* __restrict__ bf,
                                              float* __restrict__ gate) {
  const int lane = threadIdx.x & 31;
  const int n = blockIdx.x * 8 + (threadIdx.x >> 5);   // one wave per row
  if (n >= N_) return;
  float s = 0.0f;
  for (int d = lane; d < D_; d += 32) s += x[(size_t)n * D_ + d] * Wf[d];
  s += __shfl_xor(s, 1, 32);
  s += __shfl_xor(s, 2, 32);
  s += __shfl_xor(s, 4, 32);
  s += __shfl_xor(s, 8, 32);
  s += __shfl_xor(s, 16, 32);
  if (lane == 0) gate[n] = 1.0f / (1.0f + __expf(-(s + bf[0])));
}

// ---------------------------------------------------------------------------
// Launch
// ---------------------------------------------------------------------------
extern "C" void kernel_launch(void* const* d_in, const int* in_sizes, int n_in,
                              void* d_out, int out_size, void* d_ws, size_t ws_size,
                              hipStream_t stream) {
  const float* x  = (const float*)d_in[0];
  const float* Wq = (const float*)d_in[1];
  const float* bq = (const float*)d_in[2];
  const float* Wk = (const float*)d_in[3];
  const float* bk = (const float*)d_in[4];
  const float* Wv = (const float*)d_in[5];
  const float* bv = (const float*)d_in[6];
  const float* Wo = (const float*)d_in[7];
  const float* bo = (const float*)d_in[8];
  const float* Wg = (const float*)d_in[9];
  const float* bg = (const float*)d_in[10];
  const float* Wf = (const float*)d_in[11];
  const float* bf = (const float*)d_in[12];

  size_t off = 0;
  auto alloc = [&](size_t bytes) -> void* {
    void* p = (char*)d_ws + off;
    off = (off + bytes + 255) & ~(size_t)255;
    return p;
  };
  bf16*  xb    = (bf16*)alloc((size_t)N_ * D_ * 2);
  bf16*  WqT   = (bf16*)alloc((size_t)D_ * D_ * 2);
  bf16*  WkT   = (bf16*)alloc((size_t)D_ * D_ * 2);
  bf16*  WvT   = (bf16*)alloc((size_t)D_ * D_ * 2);
  bf16*  WoT   = (bf16*)alloc((size_t)D_ * D_ * 2);
  bf16*  Qb    = (bf16*)alloc((size_t)N_ * D_ * 2);
  bf16*  Kb    = (bf16*)alloc((size_t)N_ * D_ * 2);
  bf16*  Vtb   = (bf16*)alloc((size_t)N_ * D_ * 2);
  bf16*  Ob    = (bf16*)alloc((size_t)N_ * D_ * 2);
  float* gctx  = (float*)alloc((size_t)B_ * D_ * 4);
  float* vglob = (float*)alloc((size_t)B_ * D_ * 4);
  float* gate  = (float*)alloc((size_t)N_ * 4);

  // 1) precision conversion / weight transposes
  k_cvt_bf16<<<(N_ * D_ + 255) / 256, 256, 0, stream>>>(x, xb, N_ * D_);
  const int tgrid = (D_ * D_ + 255) / 256;
  k_cvt_transpose<<<tgrid, 256, 0, stream>>>(Wq, WqT, D_, D_);
  k_cvt_transpose<<<tgrid, 256, 0, stream>>>(Wk, WkT, D_, D_);
  k_cvt_transpose<<<tgrid, 256, 0, stream>>>(Wv, WvT, D_, D_);
  k_cvt_transpose<<<tgrid, 256, 0, stream>>>(Wo, WoT, D_, D_);

  // 2) Q/K/V projections (WMMA + async LDS staging), V pre-transposed per head
  const int gemm_blocks = (N_ / 128) * (D_ / 128);     // 256
  k_gemm<<<gemm_blocks, 256, 0, stream>>>(xb, WqT, bq, Qb,  N_, D_, D_, 1, nullptr, nullptr);
  k_gemm<<<gemm_blocks, 256, 0, stream>>>(xb, WkT, bk, Kb,  N_, D_, D_, 1, nullptr, nullptr);
  k_gemm<<<gemm_blocks, 256, 0, stream>>>(xb, WvT, bv, Vtb, N_, D_, D_, 2, nullptr, nullptr);

  // 3) flash attention (WMMA + async K/V staging), writes O row-major [N, D]
  const int fa_blocks = B_ * H_ * (L_ / 128);          // 512
  k_flash<<<fa_blocks, 256, 0, stream>>>(Qb, Kb, Vtb, Ob);

  // 4) global context / gate (fp32)
  k_colmean<<<(B_ * D_ + 255) / 256, 256, 0, stream>>>(x, gctx);
  k_vglobal<<<(B_ * D_ + 255) / 256, 256, 0, stream>>>(gctx, Wg, bg, vglob);
  k_gate<<<(N_ + 7) / 8, 256, 0, stream>>>(x, Wf, bf, gate);

  // 5) output projection with fused sigmoid-gated blend (WMMA, fp32 out)
  k_gemm<<<gemm_blocks, 256, 0, stream>>>(Ob, WoT, bo, d_out, N_, D_, D_, 3, gate, vglob);
}
